// Mamba_65360812310684
// MI455X (gfx1250) — matrix operand order repaired and checked by reading, not compiled, not run on hardware
//
#include <hip/hip_runtime.h>
#include <hip/hip_bf16.h>
#include <cstdint>

// ---------------------------------------------------------------------------
// Types for WMMA fragments (raw ext vectors; avoid HIP class-type vectors)
// ---------------------------------------------------------------------------
typedef unsigned int  u32x4   __attribute__((ext_vector_type(4)));
typedef __bf16        bf16x16 __attribute__((ext_vector_type(16)));
typedef float         f32x8   __attribute__((ext_vector_type(8)));

union FragBf16 { bf16x16 v; u32x4 q[2]; };

// Async global->LDS DMA types (signature learned from round-2 diagnostic:
// param0 is 'int __vector_size(16) __device__ *', param1 the AS(3) variant).
typedef int vsi4 __attribute__((vector_size(16)));
typedef __attribute__((address_space(1))) vsi4 as1_vsi4;
typedef __attribute__((address_space(3))) vsi4 as3_vsi4;

#if __has_builtin(__builtin_amdgcn_global_load_async_to_lds_b128) && \
    __has_builtin(__builtin_amdgcn_s_wait_asynccnt)
#define HAVE_ASYNC_LDS 1
#else
#define HAVE_ASYNC_LDS 0
#endif

static __device__ __forceinline__ unsigned short f32_to_bf16_bits(float f) {
    unsigned int u = __float_as_uint(f);
    unsigned int r = (u + 0x7FFFu + ((u >> 16) & 1u)) >> 16;  // RNE
    return (unsigned short)r;
}

static __device__ __forceinline__ float silu(float x) {
    return x / (1.0f + __expf(-x));
}

// ---------------------------------------------------------------------------
// TDM compile probe (never launched; shows tensor_load_to_lds path).
// ---------------------------------------------------------------------------
#if __has_builtin(__builtin_amdgcn_tensor_load_to_lds) && \
    __has_builtin(__builtin_amdgcn_s_wait_tensorcnt)
typedef unsigned int tdm_u32x4 __attribute__((ext_vector_type(4)));
typedef int          tdm_i32x4 __attribute__((ext_vector_type(4)));
typedef int          tdm_i32x8 __attribute__((ext_vector_type(8)));
__global__ void k_tdm_probe() {
    tdm_u32x4 g0 = (tdm_u32x4){0, 0, 0, 0};
    tdm_i32x8 g1 = (tdm_i32x8){0, 0, 0, 0, 0, 0, 0, 0};
    tdm_i32x4 g2 = (tdm_i32x4){0, 0, 0, 0};
    tdm_i32x4 g3 = (tdm_i32x4){0, 0, 0, 0};
#if __has_include(<hip/amd_detail/amd_gfx1250_TDM.h>)
    __builtin_amdgcn_tensor_load_to_lds(g0, g1, g2, g3, g1, 0);   // 6-arg
#else
    __builtin_amdgcn_tensor_load_to_lds(g0, g1, g2, g3, 0);       // 5-arg
#endif
    __builtin_amdgcn_s_wait_tensorcnt(0);
}
#endif

// ---------------------------------------------------------------------------
// Weight convert + transpose, output-indexed with zero padding:
//   in f32 [K,N] -> out bf16 [Npad,K]; rows n >= N are zero.
// ---------------------------------------------------------------------------
__global__ void k_transpose_pad_bf16(const float* __restrict__ in,
                                     unsigned short* __restrict__ out,
                                     int K, int N, int Npad) {
    int idx = blockIdx.x * 256 + threadIdx.x;
    if (idx >= Npad * K) return;
    int n = idx / K;
    int k = idx - n * K;
    out[idx] = (n < N) ? f32_to_bf16_bits(in[(size_t)k * N + n])
                       : (unsigned short)0;
}

// ---------------------------------------------------------------------------
// Embedding gather: x[t,:] = embed[ids[t],:]
// ---------------------------------------------------------------------------
__global__ void k_embed(const int* __restrict__ ids,
                        const float* __restrict__ embed,
                        float* __restrict__ x) {
    int t = blockIdx.x;
    int id = ids[t];
    const float* src = embed + (size_t)id * 1024;
    float* dst = x + (size_t)t * 1024;
    for (int j = threadIdx.x; j < 1024; j += blockDim.x) dst[j] = src[j];
}

// ---------------------------------------------------------------------------
// RMSNorm over D, output bf16 (GEMM A operand)
// ---------------------------------------------------------------------------
__global__ __launch_bounds__(256)
void k_rmsnorm_bf16(const float* __restrict__ x, const float* __restrict__ w,
                    unsigned short* __restrict__ out, int D) {
    __shared__ float red[256];
    int t = blockIdx.x;
    const float* row = x + (size_t)t * D;
    float s = 0.f;
    for (int j = threadIdx.x; j < D; j += 256) { float v = row[j]; s += v * v; }
    red[threadIdx.x] = s;
    __syncthreads();
    for (int st = 128; st > 0; st >>= 1) {
        if (threadIdx.x < st) red[threadIdx.x] += red[threadIdx.x + st];
        __syncthreads();
    }
    float rs = rsqrtf(red[0] / (float)D + 1e-5f);
    unsigned short* orow = out + (size_t)t * D;
    for (int j = threadIdx.x; j < D; j += 256)
        orow[j] = f32_to_bf16_bits(row[j] * rs * w[j]);
}

// ---------------------------------------------------------------------------
// WMMA bf16 GEMM:  C[M,N](f32) = A[M,K](bf16,row) x Bt[N,K](bf16,row)
// Requires: M % 256 == 0, N % 64 == 0, K % 32 == 0 (caller pads).
// MODE 0: store; MODE 1: softplus(acc + bias[n]); MODE 2: C += acc.
// Block = 256 threads = 8 waves; block tile 256x64; wave = two 16-row strips.
// B tile (64 cols x 32 K = 4KB) is staged in LDS via async DMA, double-
// buffered: the global->LDS copy of step k+1 overlaps the WMMAs of step k.
// ---------------------------------------------------------------------------
template <int MODE>
__global__ __launch_bounds__(256)
void k_wmma_gemm(const unsigned short* __restrict__ A,
                 const unsigned short* __restrict__ Bt,
                 float* __restrict__ C,
                 const float* __restrict__ bias,
                 int M, int N, int K)
{
    const int tid  = threadIdx.x;
    const int lane = tid & 31;
    const int wave = tid >> 5;
    const int row0 = blockIdx.x * 256 + wave * 32;   // strip0; strip1 = +16
    const int col  = blockIdx.y * 64;
    const int hk   = lane >> 4;    // half-wave selects K-octet group
    const int l16  = lane & 15;    // A: row M; B/C: column N

    f32x8 acc0[4], acc1[4];
#pragma unroll
    for (int t = 0; t < 4; ++t) {
        if (MODE == 2) {
#pragma unroll
            for (int r = 0; r < 8; ++r) {
                int n = col + t * 16 + l16;
                acc0[t][r] = C[(size_t)(row0 + r + hk * 8) * N + n];
                acc1[t][r] = C[(size_t)(row0 + 16 + r + hk * 8) * N + n];
            }
        } else {
#pragma unroll
            for (int r = 0; r < 8; ++r) { acc0[t][r] = 0.f; acc1[t][r] = 0.f; }
        }
    }

    const unsigned short* Arow0 = A + (size_t)(row0 + l16) * K;
    const unsigned short* Arow1 = Arow0 + (size_t)16 * K;

#if HAVE_ASYNC_LDS
    // LDS B tile, double buffered; row stride 40 elems (80B) spreads banks.
    __shared__ unsigned short lds_b[2][64 * 40];
    const int n_l = tid >> 2;                       // 0..63: B column
    const int ch  = tid & 3;                        // 16B chunk within row
    const unsigned short* gsrc = Bt + (size_t)(col + n_l) * K + ch * 8;
    const int stage = n_l * 40 + ch * 8;

    // Preload buffer 0 (kk = 0).
    __builtin_amdgcn_global_load_async_to_lds_b128(
        (as1_vsi4*)(gsrc), (as3_vsi4*)(&lds_b[0][stage]), 0, 0);

#pragma unroll 1
    for (int kk = 0; kk < K; kk += 32) {
        const int cur = (kk >> 5) & 1;
        // Stage next K-slab into the other buffer (dead issue on last iter;
        // trailing workspace pad covers the over-read).
        __builtin_amdgcn_global_load_async_to_lds_b128(
            (as1_vsi4*)(gsrc + kk + 32),
            (as3_vsi4*)(&lds_b[cur ^ 1][stage]), 0, 0);
        __builtin_amdgcn_s_wait_asynccnt(1);   // current buffer's DMA done
        __syncthreads();                       // visible to all 8 waves

        FragBf16 a0, a1;
        a0.q[0] = *(const u32x4*)(Arow0 + kk + hk * 8);
        a0.q[1] = *(const u32x4*)(Arow0 + kk + 16 + hk * 8);
        a1.q[0] = *(const u32x4*)(Arow1 + kk + hk * 8);
        a1.q[1] = *(const u32x4*)(Arow1 + kk + 16 + hk * 8);
        __builtin_prefetch(Arow0 + kk + 128, 0, 1);
        __builtin_prefetch(Arow1 + kk + 128, 0, 1);

#pragma unroll
        for (int t = 0; t < 4; ++t) {
            const unsigned short* bp = &lds_b[cur][(t * 16 + l16) * 40 + hk * 8];
            FragBf16 b;
            b.q[0] = *(const u32x4*)(bp);
            b.q[1] = *(const u32x4*)(bp + 16);
            acc0[t] = __builtin_amdgcn_wmma_f32_16x16x32_bf16(
                false, a0.v, false, b.v, (short)0, acc0[t], false, false);
            acc1[t] = __builtin_amdgcn_wmma_f32_16x16x32_bf16(
                false, a1.v, false, b.v, (short)0, acc1[t], false, false);
        }
        __syncthreads();                       // done reading lds_b[cur]
    }
#else
    const unsigned short* Bbase = Bt + (size_t)(col + l16) * K + hk * 8;
#pragma unroll 1
    for (int kk = 0; kk < K; kk += 32) {
        FragBf16 a0, a1;
        a0.q[0] = *(const u32x4*)(Arow0 + kk + hk * 8);
        a0.q[1] = *(const u32x4*)(Arow0 + kk + 16 + hk * 8);
        a1.q[0] = *(const u32x4*)(Arow1 + kk + hk * 8);
        a1.q[1] = *(const u32x4*)(Arow1 + kk + 16 + hk * 8);
        __builtin_prefetch(Arow0 + kk + 128, 0, 1);
        __builtin_prefetch(Arow1 + kk + 128, 0, 1);
#pragma unroll
        for (int t = 0; t < 4; ++t) {
            const unsigned short* bp = Bbase + (size_t)t * 16 * K + kk;
            FragBf16 b;
            b.q[0] = *(const u32x4*)(bp);
            b.q[1] = *(const u32x4*)(bp + 16);
            acc0[t] = __builtin_amdgcn_wmma_f32_16x16x32_bf16(
                false, a0.v, false, b.v, (short)0, acc0[t], false, false);
            acc1[t] = __builtin_amdgcn_wmma_f32_16x16x32_bf16(
                false, a1.v, false, b.v, (short)0, acc1[t], false, false);
        }
    }
#endif

#pragma unroll
    for (int t = 0; t < 4; ++t) {
        int n = col + t * 16 + l16;
        float bv = (MODE == 1) ? bias[n] : 0.f;
#pragma unroll
        for (int r = 0; r < 8; ++r) {
            float v0 = acc0[t][r];
            float v1 = acc1[t][r];
            if (MODE == 1) {
                v0 += bv; v0 = (v0 > 20.f) ? v0 : log1pf(__expf(v0));
                v1 += bv; v1 = (v1 > 20.f) ? v1 : log1pf(__expf(v1));
            }
            C[(size_t)(row0 + r + hk * 8) * N + n] = v0;
            C[(size_t)(row0 + 16 + r + hk * 8) * N + n] = v1;
        }
    }
}

// ---------------------------------------------------------------------------
// Depthwise causal conv (D_CONV=4) + SiLU on xi half of xz; outputs f32 + bf16
// ---------------------------------------------------------------------------
__global__ void k_conv_silu(const float* __restrict__ xz,
                            const float* __restrict__ cw,   // [2048,4]
                            const float* __restrict__ cb,   // [2048]
                            float* __restrict__ xi,
                            unsigned short* __restrict__ xib) {
    int idx = blockIdx.x * 256 + threadIdx.x;   // t*2048 + d
    int t = idx >> 11;
    int d = idx & 2047;
    int l = t & 511;
    float acc = cb[d];
#pragma unroll
    for (int w = 0; w < 4; ++w) {
        int dl = l - 3 + w;
        if (dl >= 0) acc += xz[(size_t)(t - 3 + w) * 4096 + d] * cw[d * 4 + w];
    }
    float y = silu(acc);
    xi[idx]  = y;
    xib[idx] = f32_to_bf16_bits(y);
}

// ---------------------------------------------------------------------------
// Extract dt_low (first 64 cols of padded proj[M,128]) as bf16 GEMM operand
// ---------------------------------------------------------------------------
__global__ void k_split_dt(const float* __restrict__ proj,
                           unsigned short* __restrict__ dtb) {
    int idx = blockIdx.x * 256 + threadIdx.x;   // t*64 + r
    int t = idx >> 6;
    int r = idx & 63;
    dtb[idx] = f32_to_bf16_bits(proj[(size_t)t * 128 + r]);
}

// ---------------------------------------------------------------------------
// Selective scan: 4096 independent (b,d) scans over L=512, 16 states each.
// Fused silu(z) gate; bf16 output feeds w_out WMMA GEMM.
// ---------------------------------------------------------------------------
__global__ __launch_bounds__(256)
void k_scan(const float* __restrict__ xi,     // [1024,2048]
            const float* __restrict__ delta,  // [1024,2048]
            const float* __restrict__ proj,   // [1024,128] (Bm @64, Cm @80)
            const float* __restrict__ xz,     // [1024,4096] (z at +2048)
            const float* __restrict__ A_log,  // [2048,16]
            const float* __restrict__ Dp,     // [2048]
            unsigned short* __restrict__ yb)  // [1024,2048]
{
    int tid = blockIdx.x * 256 + threadIdx.x;  // 0..4095
    int b = tid >> 11;
    int d = tid & 2047;
    float Aexp[16], h[16];
#pragma unroll
    for (int n = 0; n < 16; ++n) {
        Aexp[n] = -__expf(A_log[d * 16 + n]);
        h[n] = 0.f;
    }
    float Dv = Dp[d];
    for (int l = 0; l < 512; ++l) {
        int t = b * 512 + l;
        float dv = delta[(size_t)t * 2048 + d];
        float u  = xi[(size_t)t * 2048 + d];
        const float* pr = proj + (size_t)t * 128;
        float y = 0.f;
#pragma unroll
        for (int n = 0; n < 16; ++n) {
            float dA = __expf(dv * Aexp[n]);
            h[n] = h[n] * dA + dv * pr[64 + n] * u;
            y += h[n] * pr[80 + n];
        }
        y += u * Dv;
        float z = xz[(size_t)t * 4096 + 2048 + d];
        y *= silu(z);
        yb[(size_t)t * 2048 + d] = f32_to_bf16_bits(y);
    }
}

// ---------------------------------------------------------------------------
// Head: rmsnorm(x[:,1,:]) -> fc/bn/leaky -> fc -> fc/bn/leaky -> fc -> out[2,2]
// ---------------------------------------------------------------------------
__global__ __launch_bounds__(256)
void k_head(const float* __restrict__ x, const float* __restrict__ norm_f,
            const float* __restrict__ w1, const float* __restrict__ b1,
            const float* __restrict__ g1, const float* __restrict__ be1,
            const float* __restrict__ w2, const float* __restrict__ b2,
            const float* __restrict__ w3, const float* __restrict__ b3,
            const float* __restrict__ g2, const float* __restrict__ be2,
            const float* __restrict__ w4, const float* __restrict__ b4,
            float* __restrict__ out)
{
    __shared__ float rep[2][1024];
    __shared__ float o1[2][512];
    __shared__ float o2[2][256];
    __shared__ float o3[2][64];
    __shared__ float red[256];
    int tid = threadIdx.x;

    for (int b = 0; b < 2; ++b) {
        const float* row = x + (size_t)(b * 512 + 1) * 1024;
        float s = 0.f;
        for (int j = tid; j < 1024; j += 256) { float v = row[j]; s += v * v; }
        red[tid] = s;
        __syncthreads();
        for (int st = 128; st > 0; st >>= 1) {
            if (tid < st) red[tid] += red[tid + st];
            __syncthreads();
        }
        float rs = rsqrtf(red[0] / 1024.f + 1e-5f);
        for (int j = tid; j < 1024; j += 256) rep[b][j] = row[j] * rs * norm_f[j];
        __syncthreads();
    }

    for (int o = tid; o < 1024; o += 256) {
        int b = o >> 9, j = o & 511;
        float s = b1[j];
        for (int k = 0; k < 1024; ++k) s += rep[b][k] * w1[k * 512 + j];
        o1[b][j] = s;
    }
    __syncthreads();

    for (int j = tid; j < 512; j += 256) {
        float a = o1[0][j], c = o1[1][j];
        float m = 0.5f * (a + c);
        float v = 0.5f * ((a - m) * (a - m) + (c - m) * (c - m));
        float r = rsqrtf(v + 1e-5f);
        float va = (a - m) * r * g1[j] + be1[j];
        float vc = (c - m) * r * g1[j] + be1[j];
        o1[0][j] = va > 0.f ? va : 0.01f * va;
        o1[1][j] = vc > 0.f ? vc : 0.01f * vc;
    }
    __syncthreads();

    for (int o = tid; o < 512; o += 256) {
        int b = o >> 8, j = o & 255;
        float s = b2[j];
        for (int k = 0; k < 512; ++k) s += o1[b][k] * w2[k * 256 + j];
        o2[b][j] = s;
    }
    __syncthreads();

    if (tid < 128) {
        int b = tid >> 6, j = tid & 63;
        float s = b3[j];
        for (int k = 0; k < 256; ++k) s += o2[b][k] * w3[k * 64 + j];
        o3[b][j] = s;
    }
    __syncthreads();

    if (tid < 64) {
        int j = tid;
        float a = o3[0][j], c = o3[1][j];
        float m = 0.5f * (a + c);
        float v = 0.5f * ((a - m) * (a - m) + (c - m) * (c - m));
        float r = rsqrtf(v + 1e-5f);
        float va = (a - m) * r * g2[j] + be2[j];
        float vc = (c - m) * r * g2[j] + be2[j];
        o3[0][j] = va > 0.f ? va : 0.01f * va;
        o3[1][j] = vc > 0.f ? vc : 0.01f * vc;
    }
    __syncthreads();

    if (tid < 4) {
        int b = tid >> 1, j = tid & 1;
        float s = b4[j];
        for (int k = 0; k < 64; ++k) s += o3[b][k] * w4[k * 2 + j];
        out[b * 2 + j] = s;
    }
}

// ---------------------------------------------------------------------------
// Host orchestration
// ---------------------------------------------------------------------------
extern "C" void kernel_launch(void* const* d_in, const int* in_sizes, int n_in,
                              void* d_out, int out_size, void* d_ws, size_t ws_size,
                              hipStream_t stream) {
    (void)in_sizes; (void)n_in; (void)out_size; (void)ws_size;

    const int*   ids    = (const int*)  d_in[0];
    const float* embed  = (const float*)d_in[1];
    const float* norm_w = (const float*)d_in[2];
    const float* w_in   = (const float*)d_in[3];
    const float* conv_w = (const float*)d_in[4];
    const float* conv_b = (const float*)d_in[5];
    const float* w_x    = (const float*)d_in[6];
    const float* w_dt   = (const float*)d_in[7];
    const float* b_dt   = (const float*)d_in[8];
    const float* A_log  = (const float*)d_in[9];
    const float* D_p    = (const float*)d_in[10];
    const float* w_out  = (const float*)d_in[11];
    const float* norm_f = (const float*)d_in[12];
    const float* w1  = (const float*)d_in[13];
    const float* b1  = (const float*)d_in[14];
    const float* g1  = (const float*)d_in[15];
    const float* be1 = (const float*)d_in[16];
    const float* w2  = (const float*)d_in[17];
    const float* b2  = (const float*)d_in[18];
    const float* w3  = (const float*)d_in[19];
    const float* b3  = (const float*)d_in[20];
    const float* g2  = (const float*)d_in[21];
    const float* be2 = (const float*)d_in[22];
    const float* w4  = (const float*)d_in[23];
    const float* b4  = (const float*)d_in[24];
    float* out = (float*)d_out;

    char* ws = (char*)d_ws;
    size_t off = 0;
    auto carve = [&](size_t bytes) -> char* {
        char* p = ws + off;
        off += (bytes + 255) & ~(size_t)255;
        return p;
    };

    const int M = 1024;   // B*L tokens
    float*          x      = (float*)         carve((size_t)M * 1024 * 4);
    unsigned short* h_bf   = (unsigned short*)carve((size_t)M * 1024 * 2);
    float*          xz     = (float*)         carve((size_t)M * 4096 * 4);
    float*          xi     = (float*)         carve((size_t)M * 2048 * 4);
    unsigned short* xi_bf  = (unsigned short*)carve((size_t)M * 2048 * 2);
    float*          proj   = (float*)         carve((size_t)M * 128 * 4);   // padded N=128
    unsigned short* dt_bf  = (unsigned short*)carve((size_t)M * 64 * 2);
    float*          delta  = (float*)         carve((size_t)M * 2048 * 4);
    unsigned short* y_bf   = (unsigned short*)carve((size_t)M * 2048 * 2);
    unsigned short* w_in_t = (unsigned short*)carve((size_t)4 * 4096 * 1024 * 2);
    unsigned short* w_x_t  = (unsigned short*)carve((size_t)4 * 128 * 2048 * 2);  // padded
    unsigned short* w_dt_t = (unsigned short*)carve((size_t)4 * 2048 * 64 * 2);
    unsigned short* w_out_t= (unsigned short*)carve((size_t)4 * 1024 * 2048 * 2);
    (void)carve((size_t)1 << 20);   // pad: absorbs dead last-iter async over-read

    // ---- weight bf16 transposes (Bt layout [Npad,K], zero-padded rows) ----
    for (int i = 0; i < 4; ++i) {
        k_transpose_pad_bf16<<<(4096 * 1024) / 256, 256, 0, stream>>>(
            w_in + (size_t)i * 1024 * 4096, w_in_t + (size_t)i * 4096 * 1024,
            1024, 4096, 4096);
        k_transpose_pad_bf16<<<(128 * 2048) / 256, 256, 0, stream>>>(
            w_x + (size_t)i * 2048 * 96, w_x_t + (size_t)i * 128 * 2048,
            2048, 96, 128);
        k_transpose_pad_bf16<<<(2048 * 64) / 256, 256, 0, stream>>>(
            w_dt + (size_t)i * 64 * 2048, w_dt_t + (size_t)i * 2048 * 64,
            64, 2048, 2048);
        k_transpose_pad_bf16<<<(1024 * 2048) / 256, 256, 0, stream>>>(
            w_out + (size_t)i * 2048 * 1024, w_out_t + (size_t)i * 1024 * 2048,
            2048, 1024, 1024);
    }

    // ---- embedding ----
    k_embed<<<M, 256, 0, stream>>>(ids, embed, x);

    // ---- layers ----
    for (int i = 0; i < 4; ++i) {
        k_rmsnorm_bf16<<<M, 256, 0, stream>>>(x, norm_w + (size_t)i * 1024, h_bf, 1024);

        // xz = h @ w_in[i] : [1024,1024] x [1024,4096]
        k_wmma_gemm<0><<<dim3(M / 256, 4096 / 64), 256, 0, stream>>>(
            h_bf, w_in_t + (size_t)i * 4096 * 1024, xz, nullptr, M, 4096, 1024);

        k_conv_silu<<<(M * 2048) / 256, 256, 0, stream>>>(
            xz, conv_w + (size_t)i * 2048 * 4, conv_b + (size_t)i * 2048, xi, xi_bf);

        // proj = xi @ w_x[i] : [1024,2048] x [2048,96->128 padded]
        k_wmma_gemm<0><<<dim3(M / 256, 128 / 64), 256, 0, stream>>>(
            xi_bf, w_x_t + (size_t)i * 128 * 2048, proj, nullptr, M, 128, 2048);

        k_split_dt<<<(M * 64) / 256, 256, 0, stream>>>(proj, dt_bf);

        // delta = softplus(dt_low @ w_dt[i] + b_dt[i]) : [1024,64] x [64,2048]
        k_wmma_gemm<1><<<dim3(M / 256, 2048 / 64), 256, 0, stream>>>(
            dt_bf, w_dt_t + (size_t)i * 2048 * 64, delta,
            b_dt + (size_t)i * 2048, M, 2048, 64);

        k_scan<<<(2 * 2048) / 256, 256, 0, stream>>>(
            xi, delta, proj, xz, A_log + (size_t)i * 2048 * 16,
            D_p + (size_t)i * 2048, y_bf);

        // x += y' @ w_out[i] : [1024,2048] x [2048,1024], accumulate
        k_wmma_gemm<2><<<dim3(M / 256, 1024 / 64), 256, 0, stream>>>(
            y_bf, w_out_t + (size_t)i * 1024 * 2048, x, nullptr, M, 1024, 2048);
    }

    // ---- head ----
    k_head<<<1, 256, 0, stream>>>(x, norm_f, w1, b1, g1, be1,
                                  w2, b2, w3, b3, g2, be2, w4, b4, out);
}
